// PointnetFPModule_5153960755819
// MI455X (gfx1250) — matrix operand order repaired and checked by reading, not compiled, not run on hardware
//
#include <hip/hip_runtime.h>
#include <hip/hip_bf16.h>

// ---------------------------------------------------------------------------
// PointNet++ Feature Propagation on MI455X (gfx1250, wave32, WMMA)
//
// Pipeline (all on `stream`):
//   K0  zero BN-stat accumulators
//   K1  convert W0/W1 f32 -> f16
//   K2  3-NN interpolation + concat -> Xcat (point-major, f16, 131072 x 512)
//   K3  WMMA GEMM0 + fused BN-stats: Y0 = W0 x Xcat, sum/sumsq via LDS+atomics
//   K4  BN + ReLU -> H (f16, 131072 x 256)
//   K5  WMMA GEMM1 + fused BN-stats
//   K6  BN + ReLU -> d_out in (B, 128, N) f32 layout
//
// Fusing the stats into the GEMM epilogue removes two full re-reads of the
// pre-BN activations (134 MB + 67 MB ~= 8.6 us at 23.3 TB/s).
// ---------------------------------------------------------------------------

typedef __attribute__((ext_vector_type(16))) _Float16 v16h;
typedef __attribute__((ext_vector_type(8)))  _Float16 v8h;
typedef __attribute__((ext_vector_type(8)))  float    v8f;
typedef __attribute__((ext_vector_type(4)))  float    v4f;

constexpr int BATCH = 8;
constexpr int NPTS  = 16384;      // unknown points per batch
constexpr int MPTS  = 1024;       // known points per batch
constexpr int C1    = 256;        // unknow_feats channels
constexpr int C2    = 256;        // known_feats channels
constexpr int CIN   = C1 + C2;    // 512 concat channels
constexpr int H0CH  = 256;        // layer-0 output channels
constexpr int H1CH  = 128;        // layer-1 output channels
constexpr int NCOL  = BATCH * NPTS;   // 131072 GEMM columns
constexpr float BN_EPS = 1e-5f;

// ---------------------------------------------------------------------------
// K0: zero small float buffer (BN stats accumulators)
// ---------------------------------------------------------------------------
__global__ void fp_zero_kernel(float* p, int n) {
    int i = blockIdx.x * blockDim.x + threadIdx.x;
    if (i < n) p[i] = 0.0f;
}

// ---------------------------------------------------------------------------
// K1: f32 -> f16 weight conversion
// ---------------------------------------------------------------------------
__global__ void fp_cvt_f16_kernel(const float* __restrict__ in,
                                  _Float16* __restrict__ out, int n) {
    int i = blockIdx.x * blockDim.x + threadIdx.x;
    if (i < n) out[i] = (_Float16)in[i];
}

// ---------------------------------------------------------------------------
// K2: 3-NN inverse-distance interpolation + concat.
//   block = 256 threads = 256 points of one batch; known xyz cached in LDS.
//   Xcat is point-major: Xcat[(b*NPTS+p)*CIN + c]  (c<256: interp, c>=256: skip)
// ---------------------------------------------------------------------------
__global__ void fp_three_nn_concat_kernel(const float* __restrict__ unknown,     // (B,N,3)
                                          const float* __restrict__ known,       // (B,M,3)
                                          const float* __restrict__ unknow_feats,// (B,C1,N)
                                          const float* __restrict__ known_feats, // (B,C2,M)
                                          _Float16* __restrict__ Xcat) {
    __shared__ float kx[MPTS], ky[MPTS], kz[MPTS];
    const int b   = blockIdx.y;
    const int p   = blockIdx.x * blockDim.x + threadIdx.x;   // point in batch
    const int tid = threadIdx.x;

    for (int i = tid; i < MPTS; i += blockDim.x) {
        const float* kp = known + ((size_t)b * MPTS + i) * 3;
        kx[i] = kp[0]; ky[i] = kp[1]; kz[i] = kp[2];
    }
    __syncthreads();

    const float* up = unknown + ((size_t)b * NPTS + p) * 3;
    const float ux = up[0], uy = up[1], uz = up[2];

    float d0 = 3.4e38f, d1 = 3.4e38f, d2 = 3.4e38f;
    int   i0 = 0, i1 = 0, i2 = 0;
    for (int m = 0; m < MPTS; ++m) {
        const float dx = ux - kx[m], dy = uy - ky[m], dz = uz - kz[m];
        const float d = dx * dx + dy * dy + dz * dz;   // squared dist, as reference
        if (d < d2) {
            if (d < d1) {
                if (d < d0) { d2 = d1; i2 = i1; d1 = d0; i1 = i0; d0 = d; i0 = m; }
                else        { d2 = d1; i2 = i1; d1 = d;  i1 = m; }
            } else          { d2 = d;  i2 = m; }
        }
    }

    const float r0 = 1.0f / (d0 + 1e-8f);
    const float r1 = 1.0f / (d1 + 1e-8f);
    const float r2 = 1.0f / (d2 + 1e-8f);
    const float rs = 1.0f / (r0 + r1 + r2);
    const float w0 = r0 * rs, w1 = r1 * rs, w2 = r2 * rs;

    _Float16* xrow = Xcat + (size_t)(b * NPTS + p) * CIN;
    const float* kfb = known_feats + (size_t)b * C2 * MPTS;
    const float* ufb = unknow_feats + (size_t)b * C1 * NPTS;
    for (int c = 0; c < C2; ++c) {
        const float* kfc = kfb + (size_t)c * MPTS;
        const float v = w0 * kfc[i0] + w1 * kfc[i1] + w2 * kfc[i2];
        xrow[c] = (_Float16)v;
    }
    for (int c = 0; c < C1; ++c) {
        xrow[C2 + c] = (_Float16)ufb[(size_t)c * NPTS + p];
    }
}

// ---------------------------------------------------------------------------
// K3/K5: WMMA GEMM with fused BN-stat accumulation.
//   Y[col][m] = sum_k A[m][k] * X[col][k]
//   A: Mout x K row-major f16 (weights), X: NCOL x K row-major f16 (point-major)
//   Wave tile 32x32 (4 accumulators), block = 8 waves = 64(M) x 128(N) tile.
//   grid.x = NCOL/128, grid.y = Mout/64, block = 256 threads.
//
// Fragment layouts used here (A and B share the contiguous K order, not the ISA table order):
//   A 16x32 f16: lane l -> row (l&15); halves 0..15 = K+((l>>4)*16)..+15
//   B 32x16 f16: lane l -> col (l&15); halves 0..15 = K+((l>>4)*16)..+15
//   D 16x16 f32: lane l -> col (l&15); float r -> row ((l>>4)*8 + r)
//
// Epilogue: each lane owns 16 channels x (its 2 column sub-tiles); partials
// accumulated into a 64-entry LDS array (ds_add_f32), then one global
// atomicAdd per channel per block.
// ---------------------------------------------------------------------------
__global__ void fp_wmma_gemm_kernel(const _Float16* __restrict__ A,
                                    const _Float16* __restrict__ X,
                                    float* __restrict__ Y,
                                    float* __restrict__ gsum,
                                    float* __restrict__ gsumsq,
                                    int Mout, int K) {
    __shared__ float sm_sum[64];
    __shared__ float sm_sq[64];

    const int tid   = threadIdx.x;
    const int lane  = tid & 31;
    const int wave  = tid >> 5;
    const int tileM = blockIdx.y * 64  + (wave >> 2) * 32;
    const int tileN = blockIdx.x * 128 + (wave & 3)  * 32;

    if (tid < 64) { sm_sum[tid] = 0.0f; sm_sq[tid] = 0.0f; }
    __syncthreads();

    const int r16 = lane & 15;
    const int kA  = (lane >> 4) << 4;    // A-fragment K sub-offset (0 / 16)
    const int kB  = (lane >> 4) << 4;    // B-fragment K sub-offset (0 / 16)

    const _Float16* pa = A + (size_t)(tileM + r16) * K + kA;
    const _Float16* pb = X + (size_t)(tileN + r16) * K + kB;
    const size_t aRow16 = (size_t)16 * K;   // 16 rows down in A
    const size_t bRow16 = (size_t)16 * K;   // 16 cols over in X

    v8f c00 = {}, c01 = {}, c10 = {}, c11 = {};

    for (int kb = 0; kb < K; kb += 32) {
        v16h a0, a1, b0, b1;
        ((v8h*)&a0)[0] = *(const v8h*)(pa);
        ((v8h*)&a0)[1] = *(const v8h*)(pa + 8);
        ((v8h*)&a1)[0] = *(const v8h*)(pa + aRow16);
        ((v8h*)&a1)[1] = *(const v8h*)(pa + aRow16 + 8);
        b0 = *(const v16h*)(pb);
        b1 = *(const v16h*)(pb + bRow16);

        // prefetch next K-slab of activations (global_prefetch_b8)
        __builtin_prefetch(pb + 64, 0, 1);

        c00 = __builtin_amdgcn_wmma_f32_16x16x32_f16(false, a0, false, b0, (short)0, c00, false, false);
        c01 = __builtin_amdgcn_wmma_f32_16x16x32_f16(false, a0, false, b1, (short)0, c01, false, false);
        c10 = __builtin_amdgcn_wmma_f32_16x16x32_f16(false, a1, false, b0, (short)0, c10, false, false);
        c11 = __builtin_amdgcn_wmma_f32_16x16x32_f16(false, a1, false, b1, (short)0, c11, false, false);

        pa += 32; pb += 32;
    }

    const int nGrp = lane & 15;
    const int mOff = (lane >> 4) << 3;

    // ---- store D tiles ----
    {
        float* p00 = Y + (size_t)(tileN +  0 + nGrp) * Mout + (tileM +  0 + mOff);
        float* p01 = Y + (size_t)(tileN + 16 + nGrp) * Mout + (tileM +  0 + mOff);
        float* p10 = Y + (size_t)(tileN +  0 + nGrp) * Mout + (tileM + 16 + mOff);
        float* p11 = Y + (size_t)(tileN + 16 + nGrp) * Mout + (tileM + 16 + mOff);
        *(v4f*)(p00)     = ((v4f*)&c00)[0];
        *(v4f*)(p00 + 4) = ((v4f*)&c00)[1];
        *(v4f*)(p01)     = ((v4f*)&c01)[0];
        *(v4f*)(p01 + 4) = ((v4f*)&c01)[1];
        *(v4f*)(p10)     = ((v4f*)&c10)[0];
        *(v4f*)(p10 + 4) = ((v4f*)&c10)[1];
        *(v4f*)(p11)     = ((v4f*)&c11)[0];
        *(v4f*)(p11 + 4) = ((v4f*)&c11)[1];
    }

    // ---- fused BN-stats: per-lane partials -> LDS atomics ----
    {
        const int mloc0 = (wave >> 2) * 32 + mOff;        // block-local channel base
        const int mloc1 = mloc0 + 16;
#pragma unroll
        for (int r = 0; r < 8; ++r) {
            const float a = c00[r], b = c01[r];
            atomicAdd(&sm_sum[mloc0 + r], a + b);
            atomicAdd(&sm_sq [mloc0 + r], a * a + b * b);
            const float e = c10[r], f = c11[r];
            atomicAdd(&sm_sum[mloc1 + r], e + f);
            atomicAdd(&sm_sq [mloc1 + r], e * e + f * f);
        }
    }
    __syncthreads();

    if (tid < 64) {
        const int ch = blockIdx.y * 64 + tid;
        atomicAdd(&gsum[ch],   sm_sum[tid]);
        atomicAdd(&gsumsq[ch], sm_sq[tid]);
    }
}

// ---------------------------------------------------------------------------
// K4: BN (train, biased var) + ReLU, repack to f16 for next GEMM (point-major)
// ---------------------------------------------------------------------------
__global__ void fp_bn_relu_f16_kernel(const float* __restrict__ Y,
                                      const float* __restrict__ sum,
                                      const float* __restrict__ sumsq,
                                      const float* __restrict__ gamma,
                                      const float* __restrict__ beta,
                                      _Float16* __restrict__ out, int C) {
    const size_t idx = (size_t)blockIdx.x * blockDim.x + threadIdx.x;
    const int c = (int)(idx & (size_t)(C - 1));
    const float inv  = 1.0f / (float)NCOL;
    const float mean = sum[c] * inv;
    const float var  = sumsq[c] * inv - mean * mean;
    const float v = (Y[idx] - mean) * rsqrtf(var + BN_EPS) * gamma[c] + beta[c];
    out[idx] = (_Float16)fmaxf(v, 0.0f);
}

// ---------------------------------------------------------------------------
// K6: BN + ReLU -> final output, scattered from point-major to (B, C, N) f32
// ---------------------------------------------------------------------------
__global__ void fp_bn_relu_out_kernel(const float* __restrict__ Y,
                                      const float* __restrict__ sum,
                                      const float* __restrict__ sumsq,
                                      const float* __restrict__ gamma,
                                      const float* __restrict__ beta,
                                      float* __restrict__ out) {
    const size_t idx = (size_t)blockIdx.x * blockDim.x + threadIdx.x; // NCOL*H1CH
    const int    c   = (int)(idx & (size_t)(H1CH - 1));
    const size_t col = idx >> 7;               // / H1CH
    const int    b   = (int)(col >> 14);       // / NPTS
    const int    n   = (int)(col & (NPTS - 1));
    const float inv  = 1.0f / (float)NCOL;
    const float mean = sum[c] * inv;
    const float var  = sumsq[c] * inv - mean * mean;
    const float v = (Y[idx] - mean) * rsqrtf(var + BN_EPS) * gamma[c] + beta[c];
    out[((size_t)b * H1CH + c) * NPTS + n] = fmaxf(v, 0.0f);
}

// ---------------------------------------------------------------------------
// Host-side orchestration
// ---------------------------------------------------------------------------
extern "C" void kernel_launch(void* const* d_in, const int* in_sizes, int n_in,
                              void* d_out, int out_size, void* d_ws, size_t ws_size,
                              hipStream_t stream) {
    (void)in_sizes; (void)n_in; (void)out_size; (void)ws_size;

    const float* unknown      = (const float*)d_in[0];
    const float* known        = (const float*)d_in[1];
    const float* unknow_feats = (const float*)d_in[2];
    const float* known_feats  = (const float*)d_in[3];
    const float* W0 = (const float*)d_in[4];
    const float* g0 = (const float*)d_in[5];
    const float* b0 = (const float*)d_in[6];
    const float* W1 = (const float*)d_in[7];
    const float* g1 = (const float*)d_in[8];
    const float* b1 = (const float*)d_in[9];
    float* out = (float*)d_out;

    // Workspace layout (Y1 aliases Xcat: Xcat dead after GEMM0)
    char* ws = (char*)d_ws;
    const size_t XCAT_BYTES = (size_t)NCOL * CIN * sizeof(_Float16);   // 134 MB
    const size_t Y0_BYTES   = (size_t)NCOL * H0CH * sizeof(float);     // 134 MB
    const size_t H_BYTES    = (size_t)NCOL * H0CH * sizeof(_Float16);  //  67 MB
    _Float16* Xcat = (_Float16*)(ws);
    float*    Y1   = (float*)(ws);                                     // alias
    float*    Y0   = (float*)(ws + XCAT_BYTES);
    _Float16* H    = (_Float16*)(ws + XCAT_BYTES + Y0_BYTES);
    _Float16* W0h  = (_Float16*)(ws + XCAT_BYTES + Y0_BYTES + H_BYTES);
    _Float16* W1h  = W0h + (size_t)H0CH * CIN;
    float*    sum0   = (float*)(W1h + (size_t)H1CH * H0CH);
    float*    sumsq0 = sum0 + H0CH;
    float*    sum1   = sumsq0 + H0CH;
    float*    sumsq1 = sum1 + H1CH;

    // K0: zero BN stat accumulators (2*(256+128) = 768 floats)
    fp_zero_kernel<<<3, 256, 0, stream>>>(sum0, 2 * (H0CH + H1CH));

    // K1: weights -> f16
    fp_cvt_f16_kernel<<<(H0CH * CIN + 255) / 256, 256, 0, stream>>>(W0, W0h, H0CH * CIN);
    fp_cvt_f16_kernel<<<(H1CH * H0CH + 255) / 256, 256, 0, stream>>>(W1, W1h, H1CH * H0CH);

    // K2: 3-NN interpolate + concat
    fp_three_nn_concat_kernel<<<dim3(NPTS / 256, BATCH), 256, 0, stream>>>(
        unknown, known, unknow_feats, known_feats, Xcat);

    // K3: GEMM0 (256 x 131072 x 512) + fused BN stats
    fp_wmma_gemm_kernel<<<dim3(NCOL / 128, H0CH / 64), 256, 0, stream>>>(
        W0h, Xcat, Y0, sum0, sumsq0, H0CH, CIN);

    // K4: BN + ReLU -> H (f16)
    fp_bn_relu_f16_kernel<<<(size_t)NCOL * H0CH / 256, 256, 0, stream>>>(
        Y0, sum0, sumsq0, g0, b0, H, H0CH);

    // K5: GEMM1 (128 x 131072 x 256) + fused BN stats
    fp_wmma_gemm_kernel<<<dim3(NCOL / 128, H1CH / 64), 256, 0, stream>>>(
        W1h, H, Y1, sum1, sumsq1, H1CH, H0CH);

    // K6: BN + ReLU -> output (B, 128, N)
    fp_bn_relu_out_kernel<<<(size_t)NCOL * H1CH / 256, 256, 0, stream>>>(
        Y1, sum1, sumsq1, g1, b1, out);
}
